// MultiHeadAttention_90185723281535
// MI455X (gfx1250) — matrix operand order repaired and checked by reading, not compiled, not run on hardware
//
#include <hip/hip_runtime.h>
#include <hip/hip_bf16.h>

// ---------------------------------------------------------------------------
// MHA forward for MI455X (gfx1250): bf16 WMMA, f32 accumulate.
// Pipeline: pack X -> bf16; pack+transpose W -> bf16 [N][K];
//           3x GEMM (Q,K,V) with async global->LDS double buffering;
//           flash attention (wave-per-qtile);
//           O-projection GEMM -> f32 out.
// ---------------------------------------------------------------------------

typedef __attribute__((ext_vector_type(16))) __bf16 v16bf;
typedef __attribute__((ext_vector_type(8)))  float  v8f;

#define D_MODEL 2048
#define SEQ     2048
#define BATCH   4
#define NHEADS  16
#define HEAD_DIM 128
#define MROWS   (BATCH * SEQ)   // 8192

union AF { v16bf v; uint4 q[2]; };

static __device__ __forceinline__ unsigned short f2bf(float f) {
  union { float f; unsigned u; } x; x.f = f;
  unsigned u = x.u;
  unsigned r = u + 0x7FFFu + ((u >> 16) & 1u);   // round-to-nearest-even
  return (unsigned short)(r >> 16);
}

// CDNA5 async DMA: copy 16 bytes global -> LDS, tracked by ASYNCcnt.
static __device__ __forceinline__ void async_cp16(unsigned lds_byte_off,
                                                  const void* gptr) {
  asm volatile("global_load_async_to_lds_b128 %0, %1, off"
               :: "v"(lds_byte_off),
                  "v"((unsigned long long)(size_t)gptr)
               : "memory");
}
static __device__ __forceinline__ void wait_async0() {
  asm volatile("s_wait_asynccnt 0" ::: "memory");
}

// ---------------------------------------------------------------------------
// Pack f32 -> bf16 (flat, 8 elems/thread, b128 in / b128 out)
// ---------------------------------------------------------------------------
__global__ __launch_bounds__(256)
void pack_bf16(const float* __restrict__ in, unsigned short* __restrict__ out)
{
  size_t i = (size_t)blockIdx.x * 256 + threadIdx.x;   // 8 elems each
  const float4* src = (const float4*)in + i * 2;
  float4 a = src[0], b = src[1];
  unsigned short tmp[8] = { f2bf(a.x), f2bf(a.y), f2bf(a.z), f2bf(a.w),
                            f2bf(b.x), f2bf(b.y), f2bf(b.z), f2bf(b.w) };
  ((uint4*)out)[i] = *(uint4*)tmp;
}

// ---------------------------------------------------------------------------
// Pack + transpose: W f32 [K=2048][N=2048] -> Wt bf16 [N][K]
// 64x64 tile through LDS; both global sides coalesced.
// ---------------------------------------------------------------------------
__global__ __launch_bounds__(256)
void packT_bf16(const float* __restrict__ W, unsigned short* __restrict__ Wt)
{
  __shared__ unsigned short sT[64][72];   // [n][k], padded
  const int n0 = blockIdx.x * 64;
  const int k0 = blockIdx.y * 64;
  const int tc = threadIdx.x & 63;
  const int tg = threadIdx.x >> 6;        // 0..3
#pragma unroll
  for (int i = 0; i < 16; i++) {
    int r = tg * 16 + i;                  // k within tile
    sT[tc][r] = f2bf(W[(size_t)(k0 + r) * D_MODEL + n0 + tc]);
  }
  __syncthreads();
#pragma unroll
  for (int i = 0; i < 16; i++) {
    int nr = tg * 16 + i;                 // n within tile
    Wt[(size_t)(n0 + nr) * D_MODEL + k0 + tc] = sT[nr][tc];
  }
}

// ---------------------------------------------------------------------------
// GEMM: Out[M,N] = A[M,K](bf16) * Bt[N,K](bf16)^T + bias[N]
//   128x128 block tile, 8 waves, 32x64 wave tile, K-step 32,
//   double-buffered LDS filled by async global->LDS DMA.
//   OUT_MODE: 0 -> f32 natural, 1 -> bf16 natural,
//             2 -> bf16 scattered to Vt[B,H,Dh,S]
// ---------------------------------------------------------------------------
template<int OUT_MODE>
__global__ __launch_bounds__(256)
void gemm_bf16(const unsigned short* __restrict__ A,
               const unsigned short* __restrict__ Bt,
               const float* __restrict__ bias, void* __restrict__ Out)
{
  __shared__ __align__(16) unsigned short sA[2][128 * 32];  // [m][k]
  __shared__ __align__(16) unsigned short sB[2][128 * 32];  // [n][k]

  const int t     = threadIdx.x;
  const int lane  = t & 31;
  const int wave  = t >> 5;
  const int waveM = wave >> 1;        // 0..3
  const int waveN = wave & 1;         // 0..1
  const int hv    = lane >> 4;
  const int l15   = lane & 15;

  const int blockN = blockIdx.x * 128;
  const int blockM = blockIdx.y * 128;

  // copy mapping: thread covers 32 bytes of each 8KB tile
  const int cpRow  = t >> 1;              // 0..127
  const int cpByte = (t & 1) * 32;        // byte offset within 64B row

  const unsigned short* Arow = A  + (size_t)(blockM + cpRow) * D_MODEL;
  const unsigned short* Brow = Bt + (size_t)(blockN + cpRow) * D_MODEL;

  v8f acc[2][4] = {};

#define STAGE(bufi, kk)                                                        \
  do {                                                                         \
    const char* ga = (const char*)(Arow + (kk)) + cpByte;                      \
    const char* gb = (const char*)(Brow + (kk)) + cpByte;                      \
    unsigned la = (unsigned)(size_t)&sA[bufi][cpRow * 32] + cpByte;            \
    unsigned lb = (unsigned)(size_t)&sB[bufi][cpRow * 32] + cpByte;            \
    async_cp16(la,      ga);                                                   \
    async_cp16(la + 16, ga + 16);                                              \
    async_cp16(lb,      gb);                                                   \
    async_cp16(lb + 16, gb + 16);                                              \
  } while (0)

  STAGE(0, 0);
  int buf = 0;
  for (int k0 = 0; k0 < D_MODEL; k0 += 32) {
    wait_async0();
    __syncthreads();
    if (k0 + 32 < D_MODEL) STAGE(buf ^ 1, k0 + 32);

    AF afrag[2], bfrag[4];
#pragma unroll
    for (int ms = 0; ms < 2; ms++) {
      int row  = waveM * 32 + ms * 16 + l15;
      int base = row * 32 + hv * 8;                     // K 0-7 / 8-15
      afrag[ms].q[0] = *(const uint4*)&sA[buf][base];
      afrag[ms].q[1] = *(const uint4*)&sA[buf][base + 16];  // K 16-23 / 24-31
    }
#pragma unroll
    for (int ns = 0; ns < 4; ns++) {
      int n    = waveN * 64 + ns * 16 + l15;
      int base = n * 32 + hv * 8;                       // K 0-7 / 8-15
      bfrag[ns].q[0] = *(const uint4*)&sB[buf][base];
      bfrag[ns].q[1] = *(const uint4*)&sB[buf][base + 16];
    }
#pragma unroll
    for (int ms = 0; ms < 2; ms++)
#pragma unroll
      for (int ns = 0; ns < 4; ns++)
        acc[ms][ns] = __builtin_amdgcn_wmma_f32_16x16x32_bf16(
            false, afrag[ms].v, false, bfrag[ns].v,
            (short)0, acc[ms][ns], false, false);
    buf ^= 1;
  }
#undef STAGE

  // ---- epilogue ---------------------------------------------------------
#pragma unroll
  for (int ns = 0; ns < 4; ns++) {
    int n = blockN + waveN * 64 + ns * 16 + l15;
    float bv = bias[n];
#pragma unroll
    for (int ms = 0; ms < 2; ms++) {
#pragma unroll
      for (int r = 0; r < 8; r++) {
        int m = blockM + waveM * 32 + ms * 16 + r + hv * 8;
        float val = acc[ms][ns][r] + bv;
        if (OUT_MODE == 0) {
          ((float*)Out)[(size_t)m * D_MODEL + n] = val;
        } else if (OUT_MODE == 1) {
          ((unsigned short*)Out)[(size_t)m * D_MODEL + n] = f2bf(val);
        } else {
          int b = m >> 11, s = m & 2047;          // m = b*SEQ + s
          int h = n >> 7,  dh = n & 127;          // n = h*128 + dh
          size_t idx = ((size_t)((b * NHEADS + h) * HEAD_DIM + dh) << 11) + s;
          ((unsigned short*)Out)[idx] = f2bf(val);
        }
      }
    }
  }
}

// ---------------------------------------------------------------------------
// Flash attention: one wave per 16-row q tile. Causal, online softmax.
//   Q,K bf16 [B,S,D]; Vt bf16 [B,H,Dh,S]; Ctx bf16 [B,S,D]
// ---------------------------------------------------------------------------
__global__ __launch_bounds__(256)
void attn_kernel(const unsigned short* __restrict__ Q,
                 const unsigned short* __restrict__ K,
                 const unsigned short* __restrict__ Vt,
                 unsigned short* __restrict__ Ctx)
{
  __shared__ __align__(16) unsigned short sP[8][16 * 32];  // per-wave P tile

  const int t    = threadIdx.x;
  const int lane = t & 31;
  const int wave = t >> 5;
  const int hv   = lane >> 4;
  const int l15  = lane & 15;

  const int gw    = blockIdx.x * 8 + wave;   // 0..511
  const int bh    = gw >> 7;                 // 0..3
  const int qTile = gw & 127;
  const int b = bh >> 4, h = bh & 15;
  const int qBase = qTile * 16;

  const unsigned short* Qb = Q + (size_t)b * SEQ * D_MODEL + (size_t)h * HEAD_DIM;
  const unsigned short* Kb = K + (size_t)b * SEQ * D_MODEL + (size_t)h * HEAD_DIM;
  const unsigned short* Vb = Vt + (size_t)(b * NHEADS + h) * HEAD_DIM * SEQ;

  AF qf[4];
#pragma unroll
  for (int c = 0; c < 4; c++) {
    const unsigned short* p = Qb + (size_t)(qBase + l15) * D_MODEL + c * 32 + hv * 8;
    qf[c].q[0] = *(const uint4*)p;
    qf[c].q[1] = *(const uint4*)(p + 16);
  }

  v8f acc[8] = {};
  float mrow[8], srow[8];
#pragma unroll
  for (int r = 0; r < 8; r++) { mrow[r] = -__builtin_inff(); srow[r] = 0.f; }

  const float scale = 0.08838834764831845f;  // 1/sqrt(128)
  const int kEnd = qBase + 16;               // exclusive causal key bound

  for (int kc = 0; kc < kEnd; kc += 32) {
    v8f s0 = {}, s1 = {};
#pragma unroll
    for (int c = 0; c < 4; c++) {
      AF kf0, kf1;
      const unsigned short* p0 = Kb + (size_t)(kc + l15) * D_MODEL + c * 32 + hv * 8;
      kf0.q[0] = *(const uint4*)p0;
      kf0.q[1] = *(const uint4*)(p0 + 16);
      const unsigned short* p1 = Kb + (size_t)(kc + 16 + l15) * D_MODEL + c * 32 + hv * 8;
      kf1.q[0] = *(const uint4*)p1;
      kf1.q[1] = *(const uint4*)(p1 + 16);
      s0 = __builtin_amdgcn_wmma_f32_16x16x32_bf16(false, qf[c].v, false, kf0.v,
                                                   (short)0, s0, false, false);
      s1 = __builtin_amdgcn_wmma_f32_16x16x32_bf16(false, qf[c].v, false, kf1.v,
                                                   (short)0, s1, false, false);
    }

#pragma unroll
    for (int r = 0; r < 8; r++) {
      int row = qBase + r + hv * 8;
      int c0 = kc + l15, c1 = c0 + 16;
      float v0 = (c0 <= row) ? s0[r] * scale : -__builtin_inff();
      float v1 = (c1 <= row) ? s1[r] * scale : -__builtin_inff();
      float tmax = fmaxf(v0, v1);
#pragma unroll
      for (int d = 1; d < 16; d <<= 1)
        tmax = fmaxf(tmax, __shfl_xor(tmax, d, 32));
      float nm = fmaxf(mrow[r], tmax);
      float p0 = __expf(v0 - nm);
      float p1 = __expf(v1 - nm);
      float corr = __expf(mrow[r] - nm);
      float psum = p0 + p1;
#pragma unroll
      for (int d = 1; d < 16; d <<= 1)
        psum += __shfl_xor(psum, d, 32);
      srow[r] = srow[r] * corr + psum;
      mrow[r] = nm;
#pragma unroll
      for (int n = 0; n < 8; n++) acc[n][r] *= corr;
      sP[wave][(r + hv * 8) * 32 + l15]      = f2bf(p0);
      sP[wave][(r + hv * 8) * 32 + 16 + l15] = f2bf(p1);
    }

    // wave-private LDS RAW; DS ops are in-order per wave, make it explicit
    asm volatile("s_wait_dscnt 0" ::: "memory");

    AF pf;
    {
      int base = l15 * 32 + hv * 8;
      pf.q[0] = *(const uint4*)&sP[wave][base];
      pf.q[1] = *(const uint4*)&sP[wave][base + 16];
    }
#pragma unroll
    for (int n = 0; n < 8; n++) {
      AF vf;
      const unsigned short* p = Vb + (size_t)(n * 16 + l15) * SEQ + kc + hv * 8;
      vf.q[0] = *(const uint4*)p;
      vf.q[1] = *(const uint4*)(p + 16);
      acc[n] = __builtin_amdgcn_wmma_f32_16x16x32_bf16(false, pf.v, false, vf.v,
                                                       (short)0, acc[n], false, false);
    }
  }

#pragma unroll
  for (int r = 0; r < 8; r++) {
    float inv = 1.0f / srow[r];
    int sIdx = qBase + r + hv * 8;
#pragma unroll
    for (int n = 0; n < 8; n++) {
      float val = acc[n][r] * inv;
      Ctx[((size_t)(b * SEQ + sIdx)) * D_MODEL + h * HEAD_DIM + n * 16 + l15] = f2bf(val);
    }
  }
}

// ---------------------------------------------------------------------------
extern "C" void kernel_launch(void* const* d_in, const int* in_sizes, int n_in,
                              void* d_out, int out_size, void* d_ws, size_t ws_size,
                              hipStream_t stream)
{
  const float* X  = (const float*)d_in[0];
  const float* Wq = (const float*)d_in[1];
  const float* bq = (const float*)d_in[2];
  const float* Wk = (const float*)d_in[3];
  const float* bk = (const float*)d_in[4];
  const float* Wv = (const float*)d_in[5];
  const float* bv = (const float*)d_in[6];
  const float* Wo = (const float*)d_in[7];
  const float* bo = (const float*)d_in[8];
  float* out = (float*)d_out;

  const size_t xElems = (size_t)MROWS * D_MODEL;     // 16.7M
  const size_t wElems = (size_t)D_MODEL * D_MODEL;   // 4.2M

  unsigned short* Xb  = (unsigned short*)d_ws;       // also reused as Ctx
  unsigned short* Wqt = Xb  + xElems;
  unsigned short* Wkt = Wqt + wElems;
  unsigned short* Wvt = Wkt + wElems;
  unsigned short* Wot = Wvt + wElems;
  unsigned short* Qb  = Wot + wElems;
  unsigned short* Kb  = Qb  + xElems;
  unsigned short* Vt  = Kb  + xElems;
  unsigned short* Ctx = Xb;                          // X dead after V GEMM

  // 1) pack activations and weights to bf16 (weights transposed to [N][K])
  pack_bf16<<<(unsigned)(xElems / 8 / 256), 256, 0, stream>>>(X, Xb);
  dim3 tg(D_MODEL / 64, D_MODEL / 64);               // 32 x 32
  packT_bf16<<<tg, 256, 0, stream>>>(Wq, Wqt);
  packT_bf16<<<tg, 256, 0, stream>>>(Wk, Wkt);
  packT_bf16<<<tg, 256, 0, stream>>>(Wv, Wvt);
  packT_bf16<<<tg, 256, 0, stream>>>(Wo, Wot);

  // 2) projections (async-DMA double-buffered WMMA GEMMs)
  dim3 gg(D_MODEL / 128, MROWS / 128);               // 16 x 64
  gemm_bf16<1><<<gg, 256, 0, stream>>>(Xb, Wqt, bq, Qb);
  gemm_bf16<1><<<gg, 256, 0, stream>>>(Xb, Wkt, bk, Kb);
  gemm_bf16<2><<<gg, 256, 0, stream>>>(Xb, Wvt, bv, Vt);

  // 3) causal flash attention
  attn_kernel<<<(MROWS / 16) / 8, 256, 0, stream>>>(Qb, Kb, Vt, Ctx);

  // 4) output projection -> f32
  gemm_bf16<0><<<gg, 256, 0, stream>>>(Ctx, Wot, bo, out);
}
